// MultiLatentAttention_71786083385706
// MI455X (gfx1250) — matrix operand order repaired and checked by reading, not compiled, not run on hardware
//
#include <hip/hip_runtime.h>
#include <hip/hip_bf16.h>

// ---------------- problem constants ----------------
constexpr int B_   = 2;
constexpr int S_   = 2048;
constexpr int DIN  = 2048;
constexpr int DOUT = 2048;
constexpr int H_   = 16;
constexpr int HD_  = 128;   // head dim (non-rope part)
constexpr int DHD_ = 64;    // rope head dim
constexpr int QR   = 1536;  // Q_RANK
constexpr int KVR  = 512;   // KV_RANK
constexpr int NT   = B_ * S_;        // total tokens = 4096
constexpr int QKD  = HD_ + DHD_;     // 192
constexpr float SCALE_ = 0.07216878364870322f; // 192^-0.5

typedef __attribute__((ext_vector_type(16))) __bf16 v16bf;
typedef __attribute__((ext_vector_type(8)))  __bf16 v8bf;
typedef __attribute__((ext_vector_type(8)))  float  v8f;

__device__ inline v16bf cat16(v8bf a, v8bf b) {
    return __builtin_shufflevector(a, b, 0,1,2,3,4,5,6,7,8,9,10,11,12,13,14,15);
}
__device__ inline v8f wmma_bf16(v16bf a, v16bf b, v8f c) {
    return __builtin_amdgcn_wmma_f32_16x16x32_bf16(false, a, false, b, (short)0, c, false, false);
}
__device__ inline v8f v8f_zero() {
    v8f z = {0.f,0.f,0.f,0.f,0.f,0.f,0.f,0.f};
    return z;
}

// ---------------- elementwise conversion ----------------
__global__ void k_cvt_bf16(const float* __restrict__ src, __bf16* __restrict__ dst, long n) {
    long i = (long)blockIdx.x * 256 + threadIdx.x;
    if (i < n) dst[i] = (__bf16)src[i];
}

// src is [K][N] f32 row-major -> dst [N][K] bf16 (transposed, N-major)
__global__ void k_transpose_cvt(const float* __restrict__ src, __bf16* __restrict__ dst,
                                int K, int N) {
    long i = (long)blockIdx.x * 256 + threadIdx.x;
    long total = (long)K * N;
    if (i >= total) return;
    long n = i / K;
    long k = i % K;
    dst[i] = (__bf16)src[k * (long)N + n];
}

// ---------------- generic WMMA GEMM: C[M,N] = A[M,K] * BT[N,K]^T + bias ----------------
// A row-major bf16 (ld=K), BT row-major bf16 [N][K] (i.e., B column-contiguous in K).
// One wave computes a 64x64 tile. OUT_BF16: 1 -> bf16 out, 0 -> f32 out.
template <int OUT_BF16>
__global__ void __launch_bounds__(128)
k_gemm(const __bf16* __restrict__ A, const __bf16* __restrict__ BT,
       const float* __restrict__ bias, void* __restrict__ Cout,
       int M, int N, int K) {
    int wid    = blockIdx.x * 4 + (threadIdx.x >> 5);
    int tilesN = N >> 6;
    int tilesM = M >> 6;
    if (wid >= tilesM * tilesN) return;
    int m0   = (wid / tilesN) << 6;
    int n0   = (wid % tilesN) << 6;
    int lane = threadIdx.x & 31;
    int l16  = lane & 15;
    int hi   = lane >> 4;

    v8f acc[4][4];
#pragma unroll
    for (int i = 0; i < 4; i++)
#pragma unroll
        for (int j = 0; j < 4; j++) acc[i][j] = v8f_zero();

    for (int k0 = 0; k0 < K; k0 += 32) {
        v16bf a[4], b[4];
#pragma unroll
        for (int i = 0; i < 4; i++) {
            const __bf16* p = A + (size_t)(m0 + i * 16 + l16) * K + k0 + hi * 8;
            v8bf c0 = *(const v8bf*)p;
            v8bf c1 = *(const v8bf*)(p + 16);
            a[i] = cat16(c0, c1);
        }
#pragma unroll
        for (int j = 0; j < 4; j++) {
            const __bf16* p = BT + (size_t)(n0 + j * 16 + l16) * K + k0 + hi * 8;
            v8bf c0 = *(const v8bf*)p;
            v8bf c1 = *(const v8bf*)(p + 16);
            b[j] = cat16(c0, c1);
        }
#pragma unroll
        for (int i = 0; i < 4; i++)
#pragma unroll
            for (int j = 0; j < 4; j++)
                acc[i][j] = wmma_bf16(a[i], b[j], acc[i][j]);
    }

#pragma unroll
    for (int j = 0; j < 4; j++) {
        int col  = n0 + j * 16 + l16;
        float bv = bias[col];
#pragma unroll
        for (int i = 0; i < 4; i++) {
#pragma unroll
            for (int r = 0; r < 8; r++) {
                int row = m0 + i * 16 + r + 8 * hi;
                float v = acc[i][j][r] + bv;
                if (OUT_BF16)
                    ((__bf16*)Cout)[(size_t)row * N + col] = (__bf16)v;
                else
                    ((float*)Cout)[(size_t)row * N + col] = v;
            }
        }
    }
}

// ---------------- RMSNorm: f32 [rows, D] -> bf16 ----------------
__global__ void k_rmsnorm(const float* __restrict__ X, const float* __restrict__ gamma,
                          __bf16* __restrict__ Y, int D) {
    int row = blockIdx.x;
    const float* x = X + (size_t)row * D;
    __shared__ float red[256];
    float s = 0.f;
    for (int i = threadIdx.x; i < D; i += 256) { float v = x[i]; s += v * v; }
    red[threadIdx.x] = s;
    __syncthreads();
    for (int off = 128; off > 0; off >>= 1) {
        if ((int)threadIdx.x < off) red[threadIdx.x] += red[threadIdx.x + off];
        __syncthreads();
    }
    float rms = rsqrtf(red[0] / (float)D + 1e-5f);
    __bf16* y = Y + (size_t)row * D;
    for (int i = threadIdx.x; i < D; i += 256)
        y[i] = (__bf16)(x[i] * rms * gamma[i]);
}

// ---------------- assemble q/k with RoPE, v transpose ----------------
// qh[b][h][s][0..191]: first 128 from q_up, last 64 = rope(dq)
__global__ void k_assemble_q(const __bf16* __restrict__ q_up, const __bf16* __restrict__ dq,
                             const float* __restrict__ cosb, const float* __restrict__ sinb,
                             __bf16* __restrict__ qh) {
    long idx   = (long)blockIdx.x * 256 + threadIdx.x;
    long total = (long)B_ * H_ * S_ * QKD;
    if (idx >= total) return;
    int  d  = (int)(idx % QKD);
    long t  = idx / QKD;
    int  s  = (int)(t % S_);
    long bh = t / S_;
    int  h  = (int)(bh % H_);
    int  b  = (int)(bh / H_);
    long tok = (long)b * S_ + s;
    float val;
    if (d < HD_) {
        val = (float)q_up[tok * DOUT + h * HD_ + d];
    } else {
        int dd   = d - HD_;
        float xv = (float)dq[tok * (H_ * DHD_) + h * DHD_ + dd];
        int pp   = (dd < 32) ? dd + 32 : dd - 32;
        float xr = (float)dq[tok * (H_ * DHD_) + h * DHD_ + pp];
        float rot = (dd < 32) ? -xr : xr;
        val = xv * cosb[(long)s * DHD_ + dd] + rot * sinb[(long)s * DHD_ + dd];
    }
    qh[idx] = (__bf16)val;
}

// kh[b][h][s][0..191]: first 128 from k_up, last 64 = rope(dk) (dk broadcast over h)
__global__ void k_assemble_k(const __bf16* __restrict__ k_up, const __bf16* __restrict__ dk,
                             const float* __restrict__ cosb, const float* __restrict__ sinb,
                             __bf16* __restrict__ kh) {
    long idx   = (long)blockIdx.x * 256 + threadIdx.x;
    long total = (long)B_ * H_ * S_ * QKD;
    if (idx >= total) return;
    int  d  = (int)(idx % QKD);
    long t  = idx / QKD;
    int  s  = (int)(t % S_);
    long bh = t / S_;
    int  h  = (int)(bh % H_);
    int  b  = (int)(bh / H_);
    long tok = (long)b * S_ + s;
    float val;
    if (d < HD_) {
        val = (float)k_up[tok * DOUT + h * HD_ + d];
    } else {
        int dd   = d - HD_;
        float xv = (float)dk[tok * DHD_ + dd];
        int pp   = (dd < 32) ? dd + 32 : dd - 32;
        float xr = (float)dk[tok * DHD_ + pp];
        float rot = (dd < 32) ? -xr : xr;
        val = xv * cosb[(long)s * DHD_ + dd] + rot * sinb[(long)s * DHD_ + dd];
    }
    kh[idx] = (__bf16)val;
}

// vT[b][h][d][s] = v[b*S+s][h*128+d]
__global__ void k_assemble_v(const __bf16* __restrict__ vsrc, __bf16* __restrict__ vT) {
    long idx   = (long)blockIdx.x * 256 + threadIdx.x;
    long total = (long)B_ * H_ * HD_ * S_;
    if (idx >= total) return;
    int  s  = (int)(idx % S_);
    long t  = idx / S_;
    int  d  = (int)(t % HD_);
    long bh = t / HD_;
    int  h  = (int)(bh % H_);
    int  b  = (int)(bh / H_);
    vT[idx] = vsrc[((long)b * S_ + s) * DOUT + h * HD_ + d];
}

// ---------------- causal flash attention ----------------
// one wave per (b, h, 16-q-row tile); 64-key blocks; online softmax;
// P goes C-layout -> LDS -> A-layout bf16 for the PV matmul.
__global__ void __launch_bounds__(128)
k_attn(const __bf16* __restrict__ qh, const __bf16* __restrict__ kh,
       const __bf16* __restrict__ vT, __bf16* __restrict__ ctx) {
    __shared__ __align__(16) __bf16 plds[4][16 * 64];
    int wslot = threadIdx.x >> 5;
    int wid   = blockIdx.x * 4 + wslot;
    int qt = wid & 127;          // S/16 = 128 q tiles
    int h  = (wid >> 7) & 15;
    int b  = wid >> 11;
    int q0 = qt << 4;
    int lane = threadIdx.x & 31;
    int l16  = lane & 15;
    int hi   = lane >> 4;

    const __bf16* qbase = qh + (size_t)(b * H_ + h) * S_ * QKD;
    const __bf16* kbase = kh + (size_t)(b * H_ + h) * S_ * QKD;
    const __bf16* vbase = vT + (size_t)(b * H_ + h) * HD_ * S_;
    __bf16* myp = plds[wslot];

    // Q fragments: 16 rows x 192 -> 6 A-fragments of K=32
    v16bf aQ[6];
#pragma unroll
    for (int f = 0; f < 6; f++) {
        const __bf16* p = qbase + (size_t)(q0 + l16) * QKD + f * 32 + hi * 8;
        v8bf c0 = *(const v8bf*)p;
        v8bf c1 = *(const v8bf*)(p + 16);
        aQ[f] = cat16(c0, c1);
    }

    v8f o[8];
#pragma unroll
    for (int t = 0; t < 8; t++) o[t] = v8f_zero();
    float m[8], l[8];
#pragma unroll
    for (int r = 0; r < 8; r++) { m[r] = -3.0e38f; l[r] = 0.f; }

    for (int kb = 0; kb < q0 + 16; kb += 64) {
        // ---- scores: 16 x 64 = 4 tiles of 16x16, K = 192 over 6 steps ----
        v8f sc[4];
#pragma unroll
        for (int j = 0; j < 4; j++) sc[j] = v8f_zero();
#pragma unroll
        for (int j = 0; j < 4; j++) {
            const __bf16* colp = kbase + (size_t)(kb + j * 16 + l16) * QKD;
#pragma unroll
            for (int f = 0; f < 6; f++) {
                const __bf16* p = colp + f * 32 + hi * 8;
                v8bf c0 = *(const v8bf*)p;
                v8bf c1 = *(const v8bf*)(p + 16);
                sc[j] = wmma_bf16(aQ[f], cat16(c0, c1), sc[j]);
            }
        }
        // ---- scale + causal mask + online softmax ----
#pragma unroll
        for (int r = 0; r < 8; r++) {
            int qi = q0 + r + 8 * hi;
            float rowmax = -3.0e38f;
#pragma unroll
            for (int j = 0; j < 4; j++) {
                float sv = sc[j][r] * SCALE_;
                int ki = kb + j * 16 + l16;
                if (ki > qi) sv = -3.0e38f;
                sc[j][r] = sv;
                rowmax = fmaxf(rowmax, sv);
            }
#pragma unroll
            for (int msk = 8; msk >= 1; msk >>= 1)
                rowmax = fmaxf(rowmax, __shfl_xor(rowmax, msk, 32));
            float mnew = fmaxf(m[r], rowmax);
            float corr = __expf(m[r] - mnew);
            float rs = 0.f;
#pragma unroll
            for (int j = 0; j < 4; j++) {
                float p = __expf(sc[j][r] - mnew);
                rs += p;
                myp[(r + 8 * hi) * 64 + j * 16 + l16] = (__bf16)p;
            }
#pragma unroll
            for (int msk = 8; msk >= 1; msk >>= 1)
                rs += __shfl_xor(rs, msk, 32);
            l[r] = l[r] * corr + rs;
            m[r] = mnew;
#pragma unroll
            for (int t = 0; t < 8; t++) o[t][r] *= corr;
        }
        asm volatile("" ::: "memory"); // keep LDS stores before loads (HW is in-order per wave)
        // ---- PV: O(16x128) += P(16x64) * V(64x128) ----
#pragma unroll
        for (int kk = 0; kk < 2; kk++) {
            const __bf16* p = myp + l16 * 64 + kk * 32 + hi * 8;
            v8bf c0 = *(const v8bf*)p;
            v8bf c1 = *(const v8bf*)(p + 16);
            v16bf aP = cat16(c0, c1);
#pragma unroll
            for (int t = 0; t < 8; t++) {
                const __bf16* vp = vbase + (size_t)(t * 16 + l16) * S_ + kb + kk * 32 + hi * 8;
                v8bf d0 = *(const v8bf*)vp;
                v8bf d1 = *(const v8bf*)(vp + 16);
                o[t] = wmma_bf16(aP, cat16(d0, d1), o[t]);
            }
        }
        asm volatile("" ::: "memory");
    }

    // ---- normalize + write ctx[b][s][h*128 + d] ----
#pragma unroll
    for (int r = 0; r < 8; r++) {
        float inv = 1.0f / l[r];
        int row = q0 + r + 8 * hi;
        __bf16* outp = ctx + ((size_t)b * S_ + row) * DOUT + h * HD_;
#pragma unroll
        for (int t = 0; t < 8; t++)
            outp[t * 16 + l16] = (__bf16)(o[t][r] * inv);
    }
}

// ---------------- launcher ----------------
extern "C" void kernel_launch(void* const* d_in, const int* in_sizes, int n_in,
                              void* d_out, int out_size, void* d_ws, size_t ws_size,
                              hipStream_t stream) {
    const float* x          = (const float*)d_in[0];
    const float* cosb       = (const float*)d_in[2];
    const float* sinb       = (const float*)d_in[3];
    const float* wq_down_w  = (const float*)d_in[4];
    const float* wq_down_b  = (const float*)d_in[5];
    const float* q_gamma    = (const float*)d_in[6];
    const float* wkv_down_w = (const float*)d_in[7];
    const float* wkv_down_b = (const float*)d_in[8];
    const float* kv_gamma   = (const float*)d_in[9];
    const float* wq_up_w    = (const float*)d_in[10];
    const float* wq_up_b    = (const float*)d_in[11];
    const float* wk_up_w    = (const float*)d_in[12];
    const float* wk_up_b    = (const float*)d_in[13];
    const float* wv_up_w    = (const float*)d_in[14];
    const float* wv_up_b    = (const float*)d_in[15];
    const float* wq_dec_w   = (const float*)d_in[16];
    const float* wq_dec_b   = (const float*)d_in[17];
    const float* wk_dec_w   = (const float*)d_in[18];
    const float* wk_dec_b   = (const float*)d_in[19];
    const float* out_w      = (const float*)d_in[20];
    const float* out_b      = (const float*)d_in[21];
    float* outp             = (float*)d_out;

    // workspace carve-up (256B aligned)
    char*  wsb = (char*)d_ws;
    size_t off = 0;
    auto alloc = [&](size_t bytes) -> void* {
        void* p = wsb + off;
        off = (off + bytes + 255) & ~(size_t)255;
        return p;
    };
    __bf16* xbf    = (__bf16*)alloc((size_t)NT * DIN * 2);
    __bf16* wqdT   = (__bf16*)alloc((size_t)QR * DIN * 2);
    __bf16* wkvdT  = (__bf16*)alloc((size_t)KVR * DIN * 2);
    __bf16* wquT   = (__bf16*)alloc((size_t)DOUT * QR * 2);
    __bf16* wkuT   = (__bf16*)alloc((size_t)DOUT * KVR * 2);
    __bf16* wvuT   = (__bf16*)alloc((size_t)DOUT * KVR * 2);
    __bf16* wqdecT = (__bf16*)alloc((size_t)(H_ * DHD_) * QR * 2);
    __bf16* wkdecT = (__bf16*)alloc((size_t)DHD_ * DIN * 2);
    __bf16* woT    = (__bf16*)alloc((size_t)DOUT * DOUT * 2);
    float*  q_pre  = (float*) alloc((size_t)NT * QR * 4);
    float*  kv_pre = (float*) alloc((size_t)NT * KVR * 4);
    __bf16* q_lat  = (__bf16*)alloc((size_t)NT * QR * 2);
    __bf16* kv_lat = (__bf16*)alloc((size_t)NT * KVR * 2);
    __bf16* q_up   = (__bf16*)alloc((size_t)NT * DOUT * 2);
    __bf16* k_up   = (__bf16*)alloc((size_t)NT * DOUT * 2);
    __bf16* v_b    = (__bf16*)alloc((size_t)NT * DOUT * 2);
    __bf16* dq_b   = (__bf16*)alloc((size_t)NT * (H_ * DHD_) * 2);
    __bf16* dk_b   = (__bf16*)alloc((size_t)NT * DHD_ * 2);
    __bf16* qhB    = (__bf16*)alloc((size_t)B_ * H_ * S_ * QKD * 2);
    __bf16* khB    = (__bf16*)alloc((size_t)B_ * H_ * S_ * QKD * 2);
    __bf16* vTB    = (__bf16*)alloc((size_t)B_ * H_ * HD_ * S_ * 2);
    __bf16* ctxB   = (__bf16*)alloc((size_t)NT * DOUT * 2);
    (void)ws_size; (void)in_sizes; (void)n_in; (void)out_size;

    auto cdiv = [](long a, long b) { return (int)((a + b - 1) / b); };
    auto gblk = [](int M, int N) { int w = (M >> 6) * (N >> 6); return (w + 3) / 4; };

    // convert inputs / weights to bf16 (weights transposed to [N][K])
    k_cvt_bf16<<<cdiv((long)NT * DIN, 256), 256, 0, stream>>>(x, xbf, (long)NT * DIN);
    k_transpose_cvt<<<cdiv((long)DIN * QR, 256), 256, 0, stream>>>(wq_down_w, wqdT, DIN, QR);
    k_transpose_cvt<<<cdiv((long)DIN * KVR, 256), 256, 0, stream>>>(wkv_down_w, wkvdT, DIN, KVR);
    k_transpose_cvt<<<cdiv((long)QR * DOUT, 256), 256, 0, stream>>>(wq_up_w, wquT, QR, DOUT);
    k_transpose_cvt<<<cdiv((long)KVR * DOUT, 256), 256, 0, stream>>>(wk_up_w, wkuT, KVR, DOUT);
    k_transpose_cvt<<<cdiv((long)KVR * DOUT, 256), 256, 0, stream>>>(wv_up_w, wvuT, KVR, DOUT);
    k_transpose_cvt<<<cdiv((long)QR * (H_ * DHD_), 256), 256, 0, stream>>>(wq_dec_w, wqdecT, QR, H_ * DHD_);
    k_transpose_cvt<<<cdiv((long)DIN * DHD_, 256), 256, 0, stream>>>(wk_dec_w, wkdecT, DIN, DHD_);
    k_transpose_cvt<<<cdiv((long)DOUT * DOUT, 256), 256, 0, stream>>>(out_w, woT, DOUT, DOUT);

    // down-projections (f32 out) + RMSNorm (bf16 out)
    k_gemm<0><<<gblk(NT, QR), 128, 0, stream>>>(xbf, wqdT, wq_down_b, q_pre, NT, QR, DIN);
    k_gemm<0><<<gblk(NT, KVR), 128, 0, stream>>>(xbf, wkvdT, wkv_down_b, kv_pre, NT, KVR, DIN);
    k_rmsnorm<<<NT, 256, 0, stream>>>(q_pre, q_gamma, q_lat, QR);
    k_rmsnorm<<<NT, 256, 0, stream>>>(kv_pre, kv_gamma, kv_lat, KVR);

    // up/dec projections (bf16 out)
    k_gemm<1><<<gblk(NT, DOUT), 128, 0, stream>>>(q_lat, wquT, wq_up_b, q_up, NT, DOUT, QR);
    k_gemm<1><<<gblk(NT, DOUT), 128, 0, stream>>>(kv_lat, wkuT, wk_up_b, k_up, NT, DOUT, KVR);
    k_gemm<1><<<gblk(NT, DOUT), 128, 0, stream>>>(kv_lat, wvuT, wv_up_b, v_b, NT, DOUT, KVR);
    k_gemm<1><<<gblk(NT, H_ * DHD_), 128, 0, stream>>>(q_lat, wqdecT, wq_dec_b, dq_b, NT, H_ * DHD_, QR);
    k_gemm<1><<<gblk(NT, DHD_), 128, 0, stream>>>(xbf, wkdecT, wk_dec_b, dk_b, NT, DHD_, DIN);

    // RoPE + layout assembly
    k_assemble_q<<<cdiv((long)B_ * H_ * S_ * QKD, 256), 256, 0, stream>>>(q_up, dq_b, cosb, sinb, qhB);
    k_assemble_k<<<cdiv((long)B_ * H_ * S_ * QKD, 256), 256, 0, stream>>>(k_up, dk_b, cosb, sinb, khB);
    k_assemble_v<<<cdiv((long)B_ * H_ * HD_ * S_, 256), 256, 0, stream>>>(v_b, vTB);

    // causal flash attention: B*H*(S/16) waves / 4 waves per block
    k_attn<<<(B_ * H_ * (S_ / 16)) / 4, 128, 0, stream>>>(qhB, khB, vTB, ctxB);

    // output projection (f32 out -> d_out)
    k_gemm<0><<<gblk(NT, DOUT), 128, 0, stream>>>(ctxB, woT, out_b, outp, NT, DOUT, DOUT);
}
